// EFLSTMCRFPhase_42356967473703
// MI455X (gfx1250) — compile-verified, matrix operand order checked by reading, not run on hardware
//
#include <hip/hip_runtime.h>
#include <hip/hip_bf16.h>
#include <math.h>

typedef __attribute__((ext_vector_type(16))) _Float16 v16h;
typedef __attribute__((ext_vector_type(8)))  float    v8f;
typedef __attribute__((ext_vector_type(8)))  unsigned int v8u;

#define BATCH 64
#define T_LEN 256
#define EMBED 1152
#define PROJ  128
#define HID   256
#define H4    1024
#define ROWS  (BATCH * T_LEN)   // 16384

// ---------------------------------------------------------------------------
// WMMA f16 16x16x32 fragment helpers (wave32).
// A (16x32, MxK): lane L: row = L%16, hi = L/16.  16 halves i=0..15, v=i/2:
//   k = (v<4?0:16) + hi*8 + (v%4)*2 + (i%2)   -> consecutive k pairs = one u32
// B (32x16, KxN) assumed mirrored: lane L holds column n = L%16, same k map.
// C/D (16x16 f32): lane L: n = L%16; VGPR v: m = v + 8*(L/16).
// ---------------------------------------------------------------------------
__device__ __forceinline__ v16h load_a_frag(const _Float16* base, int ldK, int lane) {
  int row = lane & 15, hi = lane >> 4;
  const _Float16* rp = base + row * ldK;
  v8u u;
#pragma unroll
  for (int v = 0; v < 8; ++v) {
    int k = ((v < 4) ? 0 : 16) + hi * 8 + (v & 3) * 2;
    u[v] = *(const unsigned int*)(rp + k);
  }
  return __builtin_bit_cast(v16h, u);
}

__device__ __forceinline__ float sigmoidf_(float x) { return 1.0f / (1.0f + __expf(-x)); }
__device__ __forceinline__ float gelu_(float x) {
  return 0.5f * x * (1.0f + erff(x * 0.70710678118654752f));
}

__device__ __forceinline__ float block_reduce_sum(float v, float* red) {
  int tid = threadIdx.x;
  red[tid] = v; __syncthreads();
  for (int off = blockDim.x >> 1; off > 0; off >>= 1) {
    if (tid < off) red[tid] += red[tid + off];
    __syncthreads();
  }
  float r = red[0]; __syncthreads();
  return r;
}

// ---------------------------------------------------------------------------
// Pack a row-major f32 weight W[N][K] (GEMM computes X @ W^T) into f16
// B-fragments: out[((nt*ktc + kt)*32 + lane)*16 + i] = W[n][k]
// ---------------------------------------------------------------------------
__global__ void pack_b_kernel(const float* __restrict__ W, _Float16* __restrict__ out,
                              int N, int K) {
  int ktc = K / 32;
  int total = (N / 16) * ktc * 32 * 16;
  int idx = blockIdx.x * blockDim.x + threadIdx.x;
  if (idx >= total) return;
  int i    = idx & 15;
  int lane = (idx >> 4) & 31;
  int rest = idx >> 9;
  int kt   = rest % ktc;
  int nt   = rest / ktc;
  int v = i >> 1, pos = i & 1, hi = lane >> 4;
  int k = kt * 32 + ((v < 4) ? 0 : 16) + hi * 8 + (v & 3) * 2 + pos;
  int n = nt * 16 + (lane & 15);
  out[idx] = (_Float16)W[n * K + k];
}

// ---------------------------------------------------------------------------
// Input LayerNorm + cast to f16.  One block per row (row = b*T + t).
// ---------------------------------------------------------------------------
__global__ void ln_cast_kernel(const float* __restrict__ emb,
                               const float* __restrict__ g, const float* __restrict__ b,
                               _Float16* __restrict__ xln) {
  __shared__ float red[256];
  int row = blockIdx.x, tid = threadIdx.x;
  const float* x = emb + (size_t)row * EMBED;
  float s = 0.0f;
  for (int i = tid; i < EMBED; i += 256) s += x[i];
  float mu = block_reduce_sum(s, red) / (float)EMBED;
  float s2 = 0.0f;
  for (int i = tid; i < EMBED; i += 256) { float d = x[i] - mu; s2 += d * d; }
  float var = block_reduce_sum(s2, red) / (float)EMBED;
  float rs = rsqrtf(var + 1e-5f);
  for (int i = tid; i < EMBED; i += 256)
    xln[(size_t)row * EMBED + i] = (_Float16)((x[i] - mu) * rs * g[i] + b[i]);
}

// ---------------------------------------------------------------------------
// Projection GEMM + bias + GELU:  xp[t][b][128] = gelu(xln @ proj_w^T + b)
// A rows are r = b*T + t; output is time-major.
// ---------------------------------------------------------------------------
__global__ void gemm_proj_kernel(const _Float16* __restrict__ A,
                                 const v16h* __restrict__ Bp,
                                 const float* __restrict__ bias,
                                 _Float16* __restrict__ xp) {
  int lane = threadIdx.x & 31, wave = threadIdx.x >> 5;
  int tile = blockIdx.x * (blockDim.x >> 5) + wave;  // 1024 mtiles * 8 ntiles
  int mt = tile >> 3, nt = tile & 7;
  const int ktc = EMBED / 32;  // 36
  v8f acc = {};
  const _Float16* abase = A + (size_t)mt * 16 * EMBED;
  const v16h* bbase = Bp + (size_t)nt * ktc * 32;
  for (int kt = 0; kt < ktc; ++kt) {
    v16h a = load_a_frag(abase + kt * 32, EMBED, lane);
    v16h bf = bbase[kt * 32 + lane];
    acc = __builtin_amdgcn_wmma_f32_16x16x32_f16(false, a, false, bf, (short)0, acc,
                                                 false, false);
  }
  int n = nt * 16 + (lane & 15), hi = lane >> 4;
  float bv = bias[n];
#pragma unroll
  for (int v = 0; v < 8; ++v) {
    int r = mt * 16 + v + 8 * hi;     // r = b*T + t
    int bb = r >> 8, tt = r & 255;
    xp[((size_t)tt * BATCH + bb) * PROJ + n] = (_Float16)gelu_(acc[v] + bv);
  }
}

// ---------------------------------------------------------------------------
// Pre-activation GEMM: pre[r][1024] = A[r][K] @ W^T + bias  (r = t*B + b)
// ---------------------------------------------------------------------------
__global__ void gemm_pre_kernel(const _Float16* __restrict__ A,
                                const v16h* __restrict__ Bp,
                                const float* __restrict__ bias,
                                _Float16* __restrict__ pre, int K) {
  int lane = threadIdx.x & 31, wave = threadIdx.x >> 5;
  int tile = blockIdx.x * (blockDim.x >> 5) + wave;  // 1024 mtiles * 64 ntiles
  int mt = tile >> 6, nt = tile & 63;
  int ktc = K / 32;
  v8f acc = {};
  const _Float16* abase = A + (size_t)mt * 16 * K;
  const v16h* bbase = Bp + (size_t)nt * ktc * 32;
  for (int kt = 0; kt < ktc; ++kt) {
    v16h a = load_a_frag(abase + kt * 32, K, lane);
    v16h bf = bbase[kt * 32 + lane];
    acc = __builtin_amdgcn_wmma_f32_16x16x32_f16(false, a, false, bf, (short)0, acc,
                                                 false, false);
  }
  int n = nt * 16 + (lane & 15), hi = lane >> 4;
  float bv = bias[n];
#pragma unroll
  for (int v = 0; v < 8; ++v) {
    int r = mt * 16 + v + 8 * hi;
    pre[(size_t)r * H4 + n] = (_Float16)(acc[v] + bv);
  }
}

// ---------------------------------------------------------------------------
// Recurrent LSTM scan. blockIdx.x = direction (0 fwd, 1 bwd). 1024 threads =
// 32 waves; each wave owns 2 (ht,mt) pairs covering all 4 gate tiles, so the
// cell state c stays resident in accumulator-mapped registers for all 256
// steps. h is double-phased through a single 32KB LDS buffer with 2 barriers.
// ---------------------------------------------------------------------------
__global__ __launch_bounds__(1024, 1) void lstm_scan_kernel(
    const _Float16* __restrict__ pre_f, const _Float16* __restrict__ pre_b,
    const v16h* __restrict__ whhp_f, const v16h* __restrict__ whhp_b,
    _Float16* __restrict__ h16out, float* __restrict__ h32out) {
  __shared__ _Float16 hbuf[BATCH * HID];  // 32 KB
  int dir = blockIdx.x;
  const _Float16* pre = dir ? pre_b : pre_f;
  const v16h* whh = dir ? whhp_b : whhp_f;
  int dirOff = dir * HID;

  int tid = threadIdx.x, lane = tid & 31, wave = tid >> 5;
  int hi = lane >> 4, col = lane & 15;
  for (int i = tid; i < BATCH * HID; i += 1024) hbuf[i] = (_Float16)0.0f;
  __syncthreads();

  int ht[2], mt[2];
  ht[0] = (wave * 2) & 15;     mt[0] = (wave * 2) >> 4;
  ht[1] = (wave * 2 + 1) & 15; mt[1] = (wave * 2 + 1) >> 4;
  v8f c[2] = {{}, {}};

  for (int s = 0; s < T_LEN; ++s) {
    int t = dir ? (T_LEN - 1 - s) : s;
    const _Float16* pre_t = pre + (size_t)t * BATCH * H4;
    v8f acc[2][4];
#pragma unroll
    for (int p = 0; p < 2; ++p) {
      int nb = ht[p] * 16 + col;
#pragma unroll
      for (int g = 0; g < 4; ++g)
#pragma unroll
        for (int v = 0; v < 8; ++v)
          acc[p][g][v] = (float)pre_t[(size_t)(mt[p] * 16 + v + 8 * hi) * H4 +
                                      g * HID + nb];
#pragma unroll
      for (int kt = 0; kt < 8; ++kt) {
        v16h a = load_a_frag(hbuf + mt[p] * 16 * HID + kt * 32, HID, lane);
#pragma unroll
        for (int g = 0; g < 4; ++g) {
          v16h bf = whh[(size_t)((g * 16 + ht[p]) * 8 + kt) * 32 + lane];
          acc[p][g] = __builtin_amdgcn_wmma_f32_16x16x32_f16(
              false, a, false, bf, (short)0, acc[p][g], false, false);
        }
      }
    }
    __syncthreads();  // all reads of hbuf done
#pragma unroll
    for (int p = 0; p < 2; ++p) {
      int nb = ht[p] * 16 + col;
#pragma unroll
      for (int v = 0; v < 8; ++v) {
        float iv = sigmoidf_(acc[p][0][v]);
        float fv = sigmoidf_(acc[p][1][v]);
        float gv = tanhf(acc[p][2][v]);
        float ov = sigmoidf_(acc[p][3][v]);
        float cn = fv * c[p][v] + iv * gv;
        c[p][v] = cn;
        float hv = ov * tanhf(cn);
        int m = mt[p] * 16 + v + 8 * hi;
        hbuf[m * HID + nb] = (_Float16)hv;
        size_t go = ((size_t)t * BATCH + m) * (2 * HID) + dirOff + nb;
        if (h16out) h16out[go] = (_Float16)hv;
        if (h32out) h32out[go] = hv;
      }
    }
    __syncthreads();  // new h visible
  }
}

// ---------------------------------------------------------------------------
// Emissions: em[b][t][tag] = h1[t][b][:] . emis_w[tag] + emis_b[tag]
// ---------------------------------------------------------------------------
__global__ void emis_kernel(const float* __restrict__ h1, const float* __restrict__ w,
                            const float* __restrict__ bias, float* __restrict__ em) {
  int idx = blockIdx.x * blockDim.x + threadIdx.x;
  if (idx >= BATCH * T_LEN * 4) return;
  int tag = idx & 3, t = (idx >> 2) & 255, b = idx >> 10;
  const float* h = h1 + ((size_t)t * BATCH + b) * 512;
  const float* wr = w + tag * 512;
  float s = bias[tag];
  for (int k = 0; k < 512; ++k) s += h[k] * wr[k];
  em[((size_t)b * T_LEN + t) * 4 + tag] = s;
}

// ---------------------------------------------------------------------------
// Viterbi decode + ed/es index extraction. One thread per batch element.
// ---------------------------------------------------------------------------
__global__ void viterbi_kernel(const float* __restrict__ em,
                               const float* __restrict__ trans,
                               int* __restrict__ ed_idx, int* __restrict__ es_idx,
                               unsigned char* __restrict__ bp_ws) {
  int b = threadIdx.x;
  if (b >= BATCH) return;
  const float* e0 = em + (size_t)b * T_LEN * 4;
  unsigned char* bp = bp_ws + (size_t)b * T_LEN * 4;
  float v[4];
  for (int j = 0; j < 4; ++j) v[j] = e0[j];
  for (int t = 1; t < T_LEN; ++t) {
    float nv[4];
    for (int j = 0; j < 4; ++j) {
      float best = v[0] + trans[0 * 4 + j]; int bi = 0;
      for (int i = 1; i < 4; ++i) {
        float sc = v[i] + trans[i * 4 + j];
        if (sc > best) { best = sc; bi = i; }
      }
      nv[j] = best + e0[t * 4 + j];
      bp[t * 4 + j] = (unsigned char)bi;
    }
    for (int j = 0; j < 4; ++j) v[j] = nv[j];
  }
  int tag = 0; float best = v[0];
  for (int j = 1; j < 4; ++j) if (v[j] > best) { best = v[j]; tag = j; }
  unsigned char tags[T_LEN];
  tags[T_LEN - 1] = (unsigned char)tag;
  for (int t = T_LEN - 1; t >= 1; --t) { tag = bp[t * 4 + tag]; tags[t - 1] = (unsigned char)tag; }
  int ed = -1, es = -1;
  for (int t = 0; t < T_LEN; ++t) {
    if (ed < 0 && tags[t] == 1) ed = t;   // ED
    if (es < 0 && tags[t] == 3) es = t;   // ES
  }
  ed_idx[b] = (ed < 0) ? 0 : ed;
  es_idx[b] = (es < 0) ? (T_LEN / 2) : es;
}

// ---------------------------------------------------------------------------
// Head MLP: gather -> LN -> 1024->256 gelu -> 256->64 gelu -> 64->1
// One block per batch element.
// ---------------------------------------------------------------------------
__global__ void head_kernel(const float* __restrict__ h1,
                            const int* __restrict__ ed_idx, const int* __restrict__ es_idx,
                            const float* __restrict__ lng, const float* __restrict__ lnb,
                            const float* __restrict__ w1, const float* __restrict__ b1,
                            const float* __restrict__ w2, const float* __restrict__ b2,
                            const float* __restrict__ w3, const float* __restrict__ b3,
                            float* __restrict__ out) {
  __shared__ float comb[1024];
  __shared__ float red[256];
  __shared__ float hs1[256];
  __shared__ float hs2[64];
  int b = blockIdx.x, tid = threadIdx.x;
  int ed = ed_idx[b], es = es_idx[b];
  const float* he = h1 + ((size_t)ed * BATCH + b) * 512;
  const float* hss = h1 + ((size_t)es * BATCH + b) * 512;
  for (int i = tid; i < 512; i += 256) { comb[i] = he[i]; comb[512 + i] = hss[i]; }
  __syncthreads();
  float s = 0.0f;
  for (int i = tid; i < 1024; i += 256) s += comb[i];
  float mu = block_reduce_sum(s, red) / 1024.0f;
  float s2 = 0.0f;
  for (int i = tid; i < 1024; i += 256) { float d = comb[i] - mu; s2 += d * d; }
  float var = block_reduce_sum(s2, red) / 1024.0f;
  float rs = rsqrtf(var + 1e-5f);
  __syncthreads();
  for (int i = tid; i < 1024; i += 256) comb[i] = (comb[i] - mu) * rs * lng[i] + lnb[i];
  __syncthreads();
  float a1 = b1[tid];
  const float* w1r = w1 + (size_t)tid * 1024;
  for (int k = 0; k < 1024; ++k) a1 += comb[k] * w1r[k];
  hs1[tid] = gelu_(a1);
  __syncthreads();
  if (tid < 64) {
    float a2 = b2[tid];
    const float* w2r = w2 + (size_t)tid * 256;
    for (int k = 0; k < 256; ++k) a2 += hs1[k] * w2r[k];
    hs2[tid] = gelu_(a2);
  }
  __syncthreads();
  if (tid == 0) {
    float a3 = b3[0];
    for (int k = 0; k < 64; ++k) a3 += hs2[k] * w3[k];
    out[b] = a3;
  }
}

// ---------------------------------------------------------------------------
extern "C" void kernel_launch(void* const* d_in, const int* in_sizes, int n_in,
                              void* d_out, int out_size, void* d_ws, size_t ws_size,
                              hipStream_t stream) {
  (void)in_sizes; (void)n_in; (void)out_size; (void)ws_size;
  const float* emb    = (const float*)d_in[0];
  const float* ln_g   = (const float*)d_in[1];
  const float* ln_b   = (const float*)d_in[2];
  const float* proj_w = (const float*)d_in[3];
  const float* proj_b = (const float*)d_in[4];
  const float* wih0f  = (const float*)d_in[5];
  const float* whh0f  = (const float*)d_in[6];
  const float* b0f    = (const float*)d_in[7];
  const float* wih0b  = (const float*)d_in[8];
  const float* whh0b  = (const float*)d_in[9];
  const float* b0b    = (const float*)d_in[10];
  const float* wih1f  = (const float*)d_in[11];
  const float* whh1f  = (const float*)d_in[12];
  const float* b1f    = (const float*)d_in[13];
  const float* wih1b  = (const float*)d_in[14];
  const float* whh1b  = (const float*)d_in[15];
  const float* b1b    = (const float*)d_in[16];
  const float* emis_w = (const float*)d_in[17];
  const float* emis_b = (const float*)d_in[18];
  const float* ln_ef_g = (const float*)d_in[19];
  const float* ln_ef_b = (const float*)d_in[20];
  const float* ef_w1  = (const float*)d_in[21];
  const float* ef_b1  = (const float*)d_in[22];
  const float* ef_w2  = (const float*)d_in[23];
  const float* ef_b2  = (const float*)d_in[24];
  const float* ef_w3  = (const float*)d_in[25];
  const float* ef_b3  = (const float*)d_in[26];
  const float* trans  = (const float*)d_in[27];

  char* ws = (char*)d_ws;
  size_t off = 0;
  auto alloc = [&](size_t bytes) { size_t r = off; off = (off + bytes + 255) & ~(size_t)255; return r; };

  size_t o_pk_proj  = alloc((size_t)8 * 36 * 512 * 2);    // 294912
  size_t o_pk_wih0f = alloc((size_t)64 * 4 * 512 * 2);
  size_t o_pk_wih0b = alloc((size_t)64 * 4 * 512 * 2);
  size_t o_pk_whh0f = alloc((size_t)64 * 8 * 512 * 2);
  size_t o_pk_whh0b = alloc((size_t)64 * 8 * 512 * 2);
  size_t o_pk_whh1f = alloc((size_t)64 * 8 * 512 * 2);
  size_t o_pk_whh1b = alloc((size_t)64 * 8 * 512 * 2);
  size_t o_pk_wih1f = alloc((size_t)64 * 16 * 512 * 2);
  size_t o_pk_wih1b = alloc((size_t)64 * 16 * 512 * 2);
  size_t o_xp    = alloc((size_t)ROWS * PROJ * 2);        // 4 MB  f16
  size_t o_h0    = alloc((size_t)ROWS * 512 * 2);         // 16 MB f16
  size_t o_h1    = alloc((size_t)ROWS * 512 * 4);         // 32 MB f32
  size_t o_emis  = alloc((size_t)BATCH * T_LEN * 4 * 4);
  size_t o_bp    = alloc((size_t)BATCH * T_LEN * 4);
  size_t o_ed    = alloc((size_t)BATCH * 4);
  size_t o_es    = alloc((size_t)BATCH * 4);
  size_t o_big   = alloc((size_t)2 * ROWS * H4 * 2);      // 64 MB: xln, then pre_f/pre_b
  size_t o_xln   = o_big;
  size_t o_pre_f = o_big;
  size_t o_pre_b = o_big + (size_t)ROWS * H4 * 2;

  _Float16* xln  = (_Float16*)(ws + o_xln);
  _Float16* xp   = (_Float16*)(ws + o_xp);
  _Float16* h0   = (_Float16*)(ws + o_h0);
  float*    h1   = (float*)(ws + o_h1);
  _Float16* preF = (_Float16*)(ws + o_pre_f);
  _Float16* preB = (_Float16*)(ws + o_pre_b);
  float*    emis = (float*)(ws + o_emis);
  unsigned char* bp = (unsigned char*)(ws + o_bp);
  int* edi = (int*)(ws + o_ed);
  int* esi = (int*)(ws + o_es);

  auto pack = [&](const float* W, size_t oo, int N, int K) {
    int total = (N / 16) * (K / 32) * 512;
    pack_b_kernel<<<(total + 255) / 256, 256, 0, stream>>>(W, (_Float16*)(ws + oo), N, K);
  };
  pack(proj_w, o_pk_proj, PROJ, EMBED);
  pack(wih0f, o_pk_wih0f, H4, PROJ);
  pack(wih0b, o_pk_wih0b, H4, PROJ);
  pack(whh0f, o_pk_whh0f, H4, HID);
  pack(whh0b, o_pk_whh0b, H4, HID);
  pack(whh1f, o_pk_whh1f, H4, HID);
  pack(whh1b, o_pk_whh1b, H4, HID);
  pack(wih1f, o_pk_wih1f, H4, 512);
  pack(wih1b, o_pk_wih1b, H4, 512);

  // 1) input LN + cast
  ln_cast_kernel<<<ROWS, 256, 0, stream>>>(emb, ln_g, ln_b, xln);

  // 2) projection GEMM (+gelu) -> xp time-major f16
  gemm_proj_kernel<<<(ROWS / 16) * 8 / 8, 256, 0, stream>>>(
      xln, (const v16h*)(ws + o_pk_proj), proj_b, xp);

  // 3) layer-0 input pre-activations (both dirs)
  gemm_pre_kernel<<<(ROWS / 16) * 64 / 8, 256, 0, stream>>>(
      xp, (const v16h*)(ws + o_pk_wih0f), b0f, preF, PROJ);
  gemm_pre_kernel<<<(ROWS / 16) * 64 / 8, 256, 0, stream>>>(
      xp, (const v16h*)(ws + o_pk_wih0b), b0b, preB, PROJ);

  // 4) layer-0 recurrent scan (fwd + bwd blocks) -> h0 (f16, [t][b][512])
  lstm_scan_kernel<<<2, 1024, 0, stream>>>(
      preF, preB, (const v16h*)(ws + o_pk_whh0f), (const v16h*)(ws + o_pk_whh0b),
      h0, (float*)nullptr);

  // 5) layer-1 input pre-activations
  gemm_pre_kernel<<<(ROWS / 16) * 64 / 8, 256, 0, stream>>>(
      h0, (const v16h*)(ws + o_pk_wih1f), b1f, preF, 512);
  gemm_pre_kernel<<<(ROWS / 16) * 64 / 8, 256, 0, stream>>>(
      h0, (const v16h*)(ws + o_pk_wih1b), b1b, preB, 512);

  // 6) layer-1 recurrent scan -> h1 (f32, [t][b][512]) == lstm_out
  lstm_scan_kernel<<<2, 1024, 0, stream>>>(
      preF, preB, (const v16h*)(ws + o_pk_whh1f), (const v16h*)(ws + o_pk_whh1b),
      (_Float16*)nullptr, h1);

  // 7) emissions
  emis_kernel<<<(BATCH * T_LEN * 4 + 255) / 256, 256, 0, stream>>>(h1, emis_w, emis_b, emis);

  // 8) Viterbi + index extraction
  viterbi_kernel<<<1, 64, 0, stream>>>(emis, trans, edi, esi, bp);

  // 9) gather + LN + MLP head
  head_kernel<<<BATCH, 256, 0, stream>>>(h1, edi, esi, ln_ef_g, ln_ef_b,
                                         ef_w1, ef_b1, ef_w2, ef_b2, ef_w3, ef_b3,
                                         (float*)d_out);
}
